// IntBlock_17428977287601
// MI455X (gfx1250) — 8-bit matrix operand order repaired and checked by reading, compiled, not run on hardware
//
#include <hip/hip_runtime.h>
#include <stdint.h>

typedef int v8i __attribute__((ext_vector_type(8)));

#define C_IN     128
#define C_OUT    128
#define IMG_H    56
#define IMG_W    56
#define N_IMG    32
#define IMG_HW   (IMG_H * IMG_W)
#define SPATIAL  (N_IMG * IMG_HW)
#define K_ITERS  18
#define K_HALF   9
#define NTILE    64
#define A_TILE_BYTES (8 * K_ITERS * 32 * 32)

__global__ __launch_bounds__(256)
void quant_x_kernel(const float* __restrict__ x, int8_t* __restrict__ xq) {
    __shared__ int8_t lt[64][132];
    const int tile = blockIdx.x;
    const int img  = tile / 49;
    const int hw0  = (tile - img * 49) * 64;
    const int tid  = threadIdx.x;

    const int p     = tid & 63;
    const int cbase = tid >> 6;
    const float* xp = x + (size_t)img * C_IN * IMG_HW + hw0 + p;
#pragma unroll
    for (int i = 0; i < 32; ++i) {
        int c = i * 4 + cbase;
        float v = xp[(size_t)c * IMG_HW];
        int q = (int)rintf(v * 32.0f);
        q = q < -128 ? -128 : (q > 127 ? 127 : q);
        lt[p][c] = (int8_t)q;
    }
    __syncthreads();

    const int wp = tid >> 2;
    const int c0 = (tid & 3) * 32;
    const int* row = (const int*)&lt[wp][0];
    int4 lo = make_int4(row[c0 / 4 + 0], row[c0 / 4 + 1], row[c0 / 4 + 2], row[c0 / 4 + 3]);
    int4 hi = make_int4(row[c0 / 4 + 4], row[c0 / 4 + 5], row[c0 / 4 + 6], row[c0 / 4 + 7]);
    int8_t* dst = xq + ((size_t)(img * IMG_HW + hw0 + wp)) * C_IN + c0;
    *(int4*)dst = lo;
    *(int4*)(dst + 16) = hi;
}

__global__ __launch_bounds__(256)
void pack_w_kernel(const float* __restrict__ w1, const float* __restrict__ w2,
                   int8_t* __restrict__ w1p, int8_t* __restrict__ w2p) {
    int idx = blockIdx.x * blockDim.x + threadIdx.x;
    if (idx >= 2 * 8 * K_ITERS * 32) return;
    int which = idx >= 8 * K_ITERS * 32;
    int t = idx - which * 8 * K_ITERS * 32;
    const float* w = which ? w2 : w1;
    int8_t* wp = which ? w2p : w1p;
    int lane = t & 31;
    int kk   = (t >> 5) % K_ITERS;
    int mt   = (t >> 5) / K_ITERS;
    int g = lane >> 4;
    int M = mt * 16 + (lane & 15);
    alignas(16) int8_t bytes[32];
#pragma unroll
    for (int v = 0; v < 8; ++v) {
#pragma unroll
        for (int j = 0; j < 4; ++j) {
            int klocal = (v & 3) * 4 + j + 16 * g + (v >> 2) * 32;
            int K = kk * 64 + klocal;
            int tap = K >> 7;
            int c   = K & 127;
            int ky = tap / 3, kx = tap - 3 * (tap / 3);
            float wf = w[(((size_t)M * C_IN + c) * 3 + ky) * 3 + kx];
            bytes[v * 4 + j] = (int8_t)(int)wf;
        }
    }
    int8_t* dst = wp + (size_t)t * 32;
    *(int4*)dst = *(const int4*)bytes;
    *(int4*)(dst + 16) = *(const int4*)(bytes + 16);
}

template <bool FINAL>
__global__ __launch_bounds__(256)
void conv_gemm_kernel(const int8_t* __restrict__ act,
                      const int8_t* __restrict__ wpack,
                      int8_t* __restrict__ act_out,
                      const float* __restrict__ x,
                      float* __restrict__ out) {
    __shared__ __align__(32) int lds_b[K_HALF][4][32][8];

    const int n_base = blockIdx.x * NTILE;
    const int tid  = threadIdx.x;
    const int wave = tid >> 5;
    const int lane = tid & 31;

    const int8_t* aptr = wpack + (size_t)(wave * K_ITERS * 32 + lane) * 32;
    v8i acc0 = {0, 0, 0, 0, 0, 0, 0, 0};
    v8i acc1 = acc0, acc2 = acc0, acc3 = acc0;

    for (int phase = 0; phase < 2; ++phase) {
        const int kbeg = phase * K_HALF;
        for (int t = tid; t < K_HALF * 4 * 32; t += 256) {
            int kk   = kbeg + (t >> 7);
            int nt   = (t >> 5) & 3;
            int ln   = t & 31;
            int pos  = n_base + nt * 16 + (ln & 15);
            int tap  = kk >> 1;
            int dy = tap / 3 - 1;
            int dx = tap - 3 * (tap / 3) - 1;
            int img = pos / IMG_HW;
            int rem = pos - img * IMG_HW;
            int h = rem / IMG_W;
            int w = rem - h * IMG_W;
            int ih = h + dy, iw = w + dx;
            int c0 = (kk & 1) * 64 + (ln >> 4) * 16;
            int4 v0 = make_int4(0, 0, 0, 0), v1 = make_int4(0, 0, 0, 0);
            if (ih >= 0 && ih < IMG_H && iw >= 0 && iw < IMG_W) {
                const int4* src = (const int4*)(act +
                    ((size_t)((img * IMG_H + ih) * IMG_W + iw)) * C_IN + c0);
                v0 = src[0];
                v1 = src[2];
            }
            int4* dst = (int4*)&lds_b[kk - kbeg][nt][ln][0];
            dst[0] = v0;
            dst[1] = v1;
        }
        __syncthreads();

#pragma unroll 3
        for (int kk = 0; kk < K_HALF; ++kk) {
            const int kg = kbeg + kk;
            if (kg + 1 < K_ITERS)
                __builtin_prefetch(aptr + (size_t)(kg + 1) * 1024, 0, 1);
            v8i afrag = *(const v8i*)(aptr + (size_t)kg * 1024);
            v8i b0 = *(const v8i*)(&lds_b[kk][0][lane][0]);
            v8i b1 = *(const v8i*)(&lds_b[kk][1][lane][0]);
            v8i b2 = *(const v8i*)(&lds_b[kk][2][lane][0]);
            v8i b3 = *(const v8i*)(&lds_b[kk][3][lane][0]);
            acc0 = __builtin_amdgcn_wmma_i32_16x16x64_iu8(true, afrag, true, b0, acc0, false, false);
            acc1 = __builtin_amdgcn_wmma_i32_16x16x64_iu8(true, afrag, true, b1, acc1, false, false);
            acc2 = __builtin_amdgcn_wmma_i32_16x16x64_iu8(true, afrag, true, b2, acc2, false, false);
            acc3 = __builtin_amdgcn_wmma_i32_16x16x64_iu8(true, afrag, true, b3, acc3, false, false);
        }
        __syncthreads();
    }

    const int mrow = (lane >> 4) * 8;
    v8i accs[4] = {acc0, acc1, acc2, acc3};
#pragma unroll
    for (int nt = 0; nt < 4; ++nt) {
        const int pos = n_base + nt * 16 + (lane & 15);
        const int img = pos / IMG_HW;
        const int rem = pos - img * IMG_HW;
        if (!FINAL) {
            uint32_t lo = 0, hi = 0;
#pragma unroll
            for (int r = 0; r < 8; ++r) {
                float y = fmaxf((float)accs[nt][r] * (1.0f / 2048.0f), 0.0f);
                int q = (int)rintf(y * 32.0f);
                q = q > 127 ? 127 : q;
                if (r < 4) lo |= ((uint32_t)q) << (8 * r);
                else       hi |= ((uint32_t)q) << (8 * (r - 4));
            }
            uint2 st; st.x = lo; st.y = hi;
            *(uint2*)(act_out + (size_t)pos * C_OUT + wave * 16 + mrow) = st;
        } else {
#pragma unroll
            for (int r = 0; r < 8; ++r) {
                int ch = wave * 16 + mrow + r;
                size_t addr = ((size_t)img * C_OUT + ch) * IMG_HW + rem;
                float f = (float)accs[nt][r] + x[addr] * 2048.0f;
                f = fminf(fmaxf(f, -2147483647.0f), 2147483647.0f);
                out[addr] = fmaxf(f * (1.0f / 2048.0f), 0.0f);
            }
        }
    }
}

extern "C" void kernel_launch(void* const* d_in, const int* in_sizes, int n_in,
                              void* d_out, int out_size, void* d_ws, size_t ws_size,
                              hipStream_t stream) {
    const float* x  = (const float*)d_in[0];
    const float* w1 = (const float*)d_in[1];
    const float* w2 = (const float*)d_in[2];
    float* out = (float*)d_out;

    int8_t* ws = (int8_t*)d_ws;
    const size_t ACT_BYTES = (size_t)SPATIAL * C_IN;
    int8_t* xq   = ws;
    int8_t* act2 = ws + ACT_BYTES;
    int8_t* w1p  = ws + 2 * ACT_BYTES;
    int8_t* w2p  = w1p + A_TILE_BYTES;

    quant_x_kernel<<<SPATIAL / 64, 256, 0, stream>>>(x, xq);
    pack_w_kernel<<<(2 * 8 * K_ITERS * 32 + 255) / 256, 256, 0, stream>>>(w1, w2, w1p, w2p);
    conv_gemm_kernel<false><<<SPATIAL / NTILE, 256, 0, stream>>>(xq, w1p, act2, nullptr, nullptr);
    conv_gemm_kernel<true ><<<SPATIAL / NTILE, 256, 0, stream>>>(act2, w2p, nullptr, x, out);
}
